// Transformer_Encoder_86681029968041
// MI455X (gfx1250) — compile-verified, matrix operand order checked by reading, not run on hardware
//
#include <hip/hip_runtime.h>
#include <hip/hip_bf16.h>

typedef _Float16 half_t;
typedef __attribute__((ext_vector_type(16))) _Float16 v16h;
typedef __attribute__((ext_vector_type(8)))  _Float16 v8h;
typedef __attribute__((ext_vector_type(8)))  float    v8f;

typedef unsigned int tdm_u32x4 __attribute__((ext_vector_type(4)));
typedef int          tdm_i32x8 __attribute__((ext_vector_type(8)));
typedef int          tdm_i32x4 __attribute__((ext_vector_type(4)));

#if __has_builtin(__builtin_amdgcn_tensor_load_to_lds) && __has_builtin(__builtin_amdgcn_s_wait_tensorcnt)
#define HAVE_TDM 1
#else
#define HAVE_TDM 0
#endif

#define B_    2
#define S_    192
#define H_    512
#define FF_   2048
#define L_    2
#define NH_   8
#define D_    64
#define MAXL_ 192

#define BM 128
#define BN 128
#define BK 32

#if HAVE_TDM
// Issue one 2D TDM tile load: global (rows x tile_k, row stride K_units
// elements of 2 bytes) -> LDS at lds_off, packed contiguously (rows*tile_k).
static __device__ inline void tdm_load_2d(unsigned lds_off, const void* gptr,
                                          unsigned tile_k, unsigned tile_rows,
                                          unsigned K_units, unsigned rows_total)
{
  unsigned long long ga = (unsigned long long)(uintptr_t)gptr;
  tdm_u32x4 g0;
  g0[0] = 1u;                                            // count=1, user mode
  g0[1] = lds_off;                                       // lds_addr
  g0[2] = (unsigned)(ga & 0xFFFFFFFFu);                  // global_addr[31:0]
  g0[3] = (unsigned)((ga >> 32) & 0x01FFFFFFu) | (2u << 30); // addr[56:32] | type=2
  tdm_i32x8 g1;
  g1[0] = (int)(1u << 16);                               // workgroup_mask=0, data_size=1 (2B)
  g1[1] = (int)((K_units & 0xFFFFu) << 16);              // tensor_dim0[15:0]
  g1[2] = (int)(((K_units >> 16) & 0xFFFFu) | ((rows_total & 0xFFFFu) << 16)); // td0 hi | td1 lo
  g1[3] = (int)(((rows_total >> 16) & 0xFFFFu) | ((tile_k & 0xFFFFu) << 16));  // td1 hi | tile_dim0
  g1[4] = (int)(tile_rows & 0xFFFFu);                    // tile_dim1 | tile_dim2=0
  g1[5] = (int)K_units;                                  // tensor_dim0_stride[31:0]
  g1[6] = 0;                                             // stride hi | dim1_stride lo
  g1[7] = 0;
  tdm_i32x4 z4 = {0, 0, 0, 0};
  tdm_i32x8 z8 = {0, 0, 0, 0, 0, 0, 0, 0};
  // 6-arg form (clang-23 / therock headers): (g0, g1, g2, g3, g4, cpol)
  __builtin_amdgcn_tensor_load_to_lds(g0, g1, z4, z4, z8, 0);
}
#endif

// ---------------------------------------------------------------------------
// Tiled WMMA GEMM:  C[M,N] = A[M,K] (f16) * Bt[N,K] (f16, pre-transposed)
//                   + bias, optional ReLU, f32 or f16 output.
// Block = 256 threads (8 waves). Block tile 128x128, K-step 32.
// Wave w (4x2 grid) computes a 32x64 strip: 2 (M) x 4 (N) accumulators.
// TDM variant stages tiles with tensor_load_to_lds + s_wait_tensorcnt.
// ---------------------------------------------------------------------------
template<bool RELU, bool OUT_F16, bool TDM>
__global__ __launch_bounds__(256)
void gemm_wmma(const half_t* __restrict__ A, const half_t* __restrict__ Bt,
               const float* __restrict__ bias,
               float* __restrict__ Cf, half_t* __restrict__ Ch,
               int M, int N, int K)
{
  __shared__ half_t As[BM][BK];   // [m][k]
  __shared__ half_t Bs[BN][BK];   // [n][k]

  const int m0   = blockIdx.y * BM;
  const int n0   = blockIdx.x * BN;
  const int tid  = threadIdx.x;
  const int wave = tid >> 5;
  const int lane = tid & 31;
  const int mn   = lane & 15;      // row (A) / col (B) within a 16-tile
  const int hs   = lane >> 4;      // K-half selector
  const int wm   = (wave & 3) * 32;
  const int wn   = (wave >> 2) * 64;

  const int srow = tid >> 2;       // 0..63 (staging row)
  const int scol = (tid & 3) * 8;  // 0,8,16,24 (staging k)

  v8f acc[2][4];
#pragma unroll
  for (int i = 0; i < 2; ++i)
#pragma unroll
    for (int j = 0; j < 4; ++j) acc[i][j] = (v8f){};

  for (int kk = 0; kk < K; kk += BK) {
    bool tdm_done = false;
#if HAVE_TDM
    if (TDM) {
      if (tid < 32) {  // one wave issues both DMA tile loads
        unsigned ldsA = (unsigned)(uintptr_t)&As[0][0];
        unsigned ldsB = (unsigned)(uintptr_t)&Bs[0][0];
        tdm_load_2d(ldsA, A  + (size_t)m0 * K + kk, BK, BM, (unsigned)K, (unsigned)M);
        tdm_load_2d(ldsB, Bt + (size_t)n0 * K + kk, BK, BN, (unsigned)K, (unsigned)N);
        __builtin_amdgcn_s_wait_tensorcnt(0);
      }
      tdm_done = true;
    }
#endif
    if (!tdm_done) {
      v8h a0 = *(const v8h*)(A  + (size_t)(m0 + srow)      * K + kk + scol);
      v8h a1 = *(const v8h*)(A  + (size_t)(m0 + 64 + srow) * K + kk + scol);
      v8h b0 = *(const v8h*)(Bt + (size_t)(n0 + srow)      * K + kk + scol);
      v8h b1 = *(const v8h*)(Bt + (size_t)(n0 + 64 + srow) * K + kk + scol);
      *(v8h*)(&As[srow][scol])      = a0;
      *(v8h*)(&As[64 + srow][scol]) = a1;
      *(v8h*)(&Bs[srow][scol])      = b0;
      *(v8h*)(&Bs[64 + srow][scol]) = b1;
      if (kk + BK < K) {  // prefetch next K-panel
        __builtin_prefetch(A  + (size_t)(m0 + srow) * K + kk + BK + scol, 0, 0);
        __builtin_prefetch(Bt + (size_t)(n0 + srow) * K + kk + BK + scol, 0, 0);
      }
    }
    __syncthreads();

    // Fragment loads per ISA layout: halves t=0..7 -> K = hs*8 + t,
    // halves t=8..15 -> K = 16 + hs*8 + (t-8). Contiguous b128 pairs.
    v16h af[2], bf[4];
#pragma unroll
    for (int i = 0; i < 2; ++i) {
      const half_t* ap = &As[wm + i * 16 + mn][hs * 8];
      v8h lo = *(const v8h*)ap, hi = *(const v8h*)(ap + 16);
#pragma unroll
      for (int t = 0; t < 8; ++t) { af[i][t] = lo[t]; af[i][t + 8] = hi[t]; }
    }
#pragma unroll
    for (int j = 0; j < 4; ++j) {
      const half_t* bp = &Bs[wn + j * 16 + mn][hs * 8];
      v8h lo = *(const v8h*)bp, hi = *(const v8h*)(bp + 16);
#pragma unroll
      for (int t = 0; t < 8; ++t) { bf[j][t] = lo[t]; bf[j][t + 8] = hi[t]; }
    }
#pragma unroll
    for (int i = 0; i < 2; ++i)
#pragma unroll
      for (int j = 0; j < 4; ++j)
        acc[i][j] = __builtin_amdgcn_wmma_f32_16x16x32_f16(
            false, af[i], false, bf[j], (short)0, acc[i][j], false, false);
    __syncthreads();
  }

  // C/D layout: vgpr v, lane l -> m = v + 8*(l>>4), n = l&15
#pragma unroll
  for (int i = 0; i < 2; ++i) {
    const int rbase = m0 + wm + i * 16 + hs * 8;
#pragma unroll
    for (int j = 0; j < 4; ++j) {
      const int c = n0 + wn + j * 16 + mn;
      const float bb = bias ? bias[c] : 0.f;
#pragma unroll
      for (int v = 0; v < 8; ++v) {
        float x = acc[i][j][v] + bb;
        if (RELU) x = fmaxf(x, 0.f);
        const size_t o = (size_t)(rbase + v) * N + c;
        if (OUT_F16) Ch[o] = (half_t)x; else Cf[o] = x;
      }
    }
  }
}

// ---------------------------------------------------------------------------
__global__ void cvt_f16(const float* __restrict__ in, half_t* __restrict__ out, int n)
{
  int i = blockIdx.x * 256 + threadIdx.x;
  if (i < n) out[i] = (half_t)in[i];
}

// f32 [K,N] -> f16 [N,K] (weight transpose for the GEMM B operand)
__global__ void cvt_f16_t(const float* __restrict__ in, half_t* __restrict__ out,
                          int K, int N)
{
  int i = blockIdx.x * 256 + threadIdx.x;
  if (i >= K * N) return;
  int k = i / N, n = i % N;
  out[(size_t)n * K + k] = (half_t)in[i];
}

// ---------------------------------------------------------------------------
// Build a chunk of pe4 rows (f16): row g=(b,i,j), 2048 cols = 4 gathered
// pe_table rows of 512 (dss, dse, des, dee).
// ---------------------------------------------------------------------------
__global__ void pe4_build(const int* __restrict__ pos_s, const int* __restrict__ pos_e,
                          const float* __restrict__ pe, half_t* __restrict__ out,
                          int row_base, int nelem)
{
  int idx = blockIdx.x * 256 + threadIdx.x;
  if (idx >= nelem) return;
  int c   = idx & 2047;
  int r   = idx >> 11;
  int g   = row_base + r;
  int j   = g % S_;
  int t   = g / S_;
  int i   = t % S_;
  int b   = t / S_;
  int seg = c >> 9;
  int col = c & 511;
  int si = pos_s[b * S_ + i], sj = pos_s[b * S_ + j];
  int ei = pos_e[b * S_ + i], ej = pos_e[b * S_ + j];
  int d;
  switch (seg) {
    case 0:  d = si - sj; break;
    case 1:  d = si - ej; break;
    case 2:  d = ei - sj; break;
    default: d = ei - ej; break;
  }
  d += MAXL_;
  out[idx] = (half_t)pe[(size_t)d * H_ + col];
}

// ---------------------------------------------------------------------------
// scores[b,h,q,k] = ((q+u)·k + (q+vp)·r) * 1/sqrt(D), masked.
// ---------------------------------------------------------------------------
__global__ void scores_kernel(const float* __restrict__ q, const float* __restrict__ kmat,
                              const half_t* __restrict__ r,
                              const float* __restrict__ u, const float* __restrict__ vp,
                              const int* __restrict__ sen_len, const int* __restrict__ lat_len,
                              float* __restrict__ scores)
{
  int idx = blockIdx.x * 256 + threadIdx.x;
  if (idx >= B_ * NH_ * S_ * S_) return;
  int kj = idx % S_;
  int t  = idx / S_;
  int qi = t % S_;  t /= S_;
  int h  = t % NH_;
  int b  = t / NH_;
  const float*  qv = q    + (size_t)(b * S_ + qi) * H_ + h * D_;
  const float*  kv = kmat + (size_t)(b * S_ + kj) * H_ + h * D_;
  const half_t* rv = r    + ((size_t)(b * S_ + qi) * S_ + kj) * H_ + h * D_;
  const float*  uu = u  + h * D_;
  const float*  vv = vp + h * D_;
  float ac = 0.f, bd = 0.f;
#pragma unroll 8
  for (int d = 0; d < D_; ++d) {
    float qd = qv[d];
    ac += (qd + uu[d]) * kv[d];
    bd += (qd + vv[d]) * (float)rv[d];
  }
  float s = (ac + bd) * 0.125f;   // 1/sqrt(64)
  if (kj >= sen_len[b] + lat_len[b]) s = -1e15f;
  scores[idx] = s;
}

// ---------------------------------------------------------------------------
// Row-wise softmax over last axis (len S_=192). One block (192 thr) per row.
// ---------------------------------------------------------------------------
__global__ void softmax_kernel(float* __restrict__ scores)
{
  __shared__ float sh[256];
  int row = blockIdx.x;
  int t   = threadIdx.x;                       // 0..191
  float* p = scores + (size_t)row * S_;
  float x = p[t];
  sh[t] = x;
  if (t < 64) sh[192 + t] = -3.0e38f;
  __syncthreads();
  for (int s = 128; s > 0; s >>= 1) { if (t < s) sh[t] = fmaxf(sh[t], sh[t + s]); __syncthreads(); }
  float m = sh[0];
  __syncthreads();
  float e = __expf(x - m);
  sh[t] = e;
  if (t < 64) sh[192 + t] = 0.f;
  __syncthreads();
  for (int s = 128; s > 0; s >>= 1) { if (t < s) sh[t] += sh[t + s]; __syncthreads(); }
  p[t] = e * (1.f / sh[0]);
}

// ---------------------------------------------------------------------------
// o[b,q,h,d] = sum_k attn[b,h,q,k] * v[b,k,h,d]
// ---------------------------------------------------------------------------
__global__ void attnv_kernel(const float* __restrict__ attn, const float* __restrict__ v,
                             float* __restrict__ o)
{
  int idx = blockIdx.x * 256 + threadIdx.x;
  if (idx >= B_ * S_ * H_) return;
  int d  = idx & 63;
  int h  = (idx >> 6) & 7;
  int t  = idx >> 9;
  int qi = t % S_;
  int b  = t / S_;
  const float* arow = attn + ((size_t)(b * NH_ + h) * S_ + qi) * S_;
  const float* vcol = v + (size_t)b * S_ * H_ + h * D_ + d;
  float acc = 0.f;
  for (int kj = 0; kj < S_; ++kj) acc += arow[kj] * vcol[(size_t)kj * H_];
  o[idx] = acc;
}

// ---------------------------------------------------------------------------
// out = LayerNorm(res + y) * g + b ; one block (256 thr) per H_=512 row.
// ---------------------------------------------------------------------------
__global__ void add_ln_kernel(const float* __restrict__ res, const float* __restrict__ y,
                              const float* __restrict__ g, const float* __restrict__ bta,
                              float* __restrict__ out)
{
  __shared__ float sh[256];
  int row = blockIdx.x;
  int t   = threadIdx.x;
  const size_t base = (size_t)row * H_;
  float v0 = res[base + t]       + y[base + t];
  float v1 = res[base + 256 + t] + y[base + 256 + t];
  sh[t] = v0 + v1;
  __syncthreads();
  for (int s = 128; s > 0; s >>= 1) { if (t < s) sh[t] += sh[t + s]; __syncthreads(); }
  float mean = sh[0] * (1.f / H_);
  __syncthreads();
  float d0 = v0 - mean, d1 = v1 - mean;
  sh[t] = d0 * d0 + d1 * d1;
  __syncthreads();
  for (int s = 128; s > 0; s >>= 1) { if (t < s) sh[t] += sh[t + s]; __syncthreads(); }
  float rstd = rsqrtf(sh[0] * (1.f / H_) + 1e-5f);
  out[base + t]       = d0 * rstd * g[t]       + bta[t];
  out[base + 256 + t] = d1 * rstd * g[t + 256] + bta[t + 256];
}

// ---------------------------------------------------------------------------
extern "C" void kernel_launch(void* const* d_in, const int* in_sizes, int n_in,
                              void* d_out, int out_size, void* d_ws, size_t ws_size,
                              hipStream_t stream)
{
  (void)in_sizes; (void)n_in; (void)out_size; (void)ws_size;

  const float* inp      = (const float*)d_in[0];
  const int*   pos_s    = (const int*)  d_in[1];
  const int*   pos_e    = (const int*)  d_in[2];
  const int*   sen_len  = (const int*)  d_in[3];
  const int*   lat_len  = (const int*)  d_in[4];
  const float* pe_table = (const float*)d_in[5];
  const float* fus_w1   = (const float*)d_in[6];
  const float* fus_b1   = (const float*)d_in[7];
  const float* fus_w2   = (const float*)d_in[8];
  const float* fus_b2   = (const float*)d_in[9];
  const float* qw = (const float*)d_in[10]; const float* qb = (const float*)d_in[11];
  const float* kw = (const float*)d_in[12]; const float* kb = (const float*)d_in[13];
  const float* vw = (const float*)d_in[14]; const float* vb = (const float*)d_in[15];
  const float* rw = (const float*)d_in[16]; const float* rb = (const float*)d_in[17];
  const float* u  = (const float*)d_in[18]; const float* vp = (const float*)d_in[19];
  const float* fw = (const float*)d_in[20]; const float* fb = (const float*)d_in[21];
  const float* ln1_g = (const float*)d_in[22]; const float* ln1_b = (const float*)d_in[23];
  const float* ffn_w1 = (const float*)d_in[24]; const float* ffn_b1 = (const float*)d_in[25];
  const float* ffn_w2 = (const float*)d_in[26]; const float* ffn_b2 = (const float*)d_in[27];
  const float* ln2_g = (const float*)d_in[28]; const float* ln2_b = (const float*)d_in[29];

  char* ws = (char*)d_ws;
  size_t off = 0;
  auto alloc = [&](size_t bytes) -> void* {
    void* p = ws + off;
    off = (off + bytes + 255) & ~(size_t)255;
    return p;
  };

  const int M  = B_ * S_;            // 384
  const int R  = B_ * S_ * S_;       // 73728 rel-pos rows
  const int CH = 9216;               // chunk rows (div by 128), 8 chunks

  // all *t weight buffers hold [N,K] (transposed) f16
  half_t* fus_w1t  = (half_t*)alloc((size_t)2048 * 2048 * 2);
  half_t* fus_w2t  = (half_t*)alloc((size_t)2048 * 512 * 2);
  half_t* qwt      = (half_t*)alloc((size_t)L_ * H_ * H_ * 2);
  half_t* kwt      = (half_t*)alloc((size_t)L_ * H_ * H_ * 2);
  half_t* vwt      = (half_t*)alloc((size_t)L_ * H_ * H_ * 2);
  half_t* rwt      = (half_t*)alloc((size_t)L_ * H_ * H_ * 2);
  half_t* fwt      = (half_t*)alloc((size_t)L_ * H_ * H_ * 2);
  half_t* ffn_w1t  = (half_t*)alloc((size_t)L_ * H_ * FF_ * 2);
  half_t* ffn_w2t  = (half_t*)alloc((size_t)L_ * FF_ * H_ * 2);
  half_t* relpos_h = (half_t*)alloc((size_t)R * H_ * 2);
  half_t* r_h      = (half_t*)alloc((size_t)R * H_ * 2);
  half_t* pe4_h    = (half_t*)alloc((size_t)CH * 4 * H_ * 2);
  half_t* h_h      = (half_t*)alloc((size_t)CH * 4 * H_ * 2);
  float*  xbuf     = (float*) alloc((size_t)M * H_ * 4);
  half_t* x_h      = (half_t*)alloc((size_t)M * H_ * 2);
  float*  qbuf     = (float*) alloc((size_t)M * H_ * 4);
  float*  kbuf     = (float*) alloc((size_t)M * H_ * 4);
  float*  vbuf     = (float*) alloc((size_t)M * H_ * 4);
  float*  scores   = (float*) alloc((size_t)B_ * NH_ * S_ * S_ * 4);
  float*  obuf     = (float*) alloc((size_t)M * H_ * 4);
  half_t* o_h      = (half_t*)alloc((size_t)M * H_ * 2);
  float*  projbuf  = (float*) alloc((size_t)M * H_ * 4);
  half_t* ff1_h    = (half_t*)alloc((size_t)M * FF_ * 2);
  float*  ff2buf   = (float*) alloc((size_t)M * H_ * 4);

  auto cvt = [&](const float* src, half_t* dst, size_t n) {
    cvt_f16<<<dim3((unsigned)((n + 255) / 256)), dim3(256), 0, stream>>>(src, dst, (int)n);
  };
  auto cvtT = [&](const float* src, half_t* dst, int K, int N) {
    size_t n = (size_t)K * N;
    cvt_f16_t<<<dim3((unsigned)((n + 255) / 256)), dim3(256), 0, stream>>>(src, dst, K, N);
  };
  auto gemm = [&](const half_t* A, const half_t* Bt, const float* bias,
                  float* Cf, half_t* Ch, int Mg, int Ng, int Kg, bool relu, bool tdm) {
    dim3 grid((unsigned)(Ng / BN), (unsigned)(Mg / BM)), blk(256);
    if (tdm) {
      if (relu) {
        if (Ch) gemm_wmma<true,  true,  true><<<grid, blk, 0, stream>>>(A, Bt, bias, Cf, Ch, Mg, Ng, Kg);
        else    gemm_wmma<true,  false, true><<<grid, blk, 0, stream>>>(A, Bt, bias, Cf, Ch, Mg, Ng, Kg);
      } else {
        if (Ch) gemm_wmma<false, true,  true><<<grid, blk, 0, stream>>>(A, Bt, bias, Cf, Ch, Mg, Ng, Kg);
        else    gemm_wmma<false, false, true><<<grid, blk, 0, stream>>>(A, Bt, bias, Cf, Ch, Mg, Ng, Kg);
      }
    } else {
      if (relu) {
        if (Ch) gemm_wmma<true,  true,  false><<<grid, blk, 0, stream>>>(A, Bt, bias, Cf, Ch, Mg, Ng, Kg);
        else    gemm_wmma<true,  false, false><<<grid, blk, 0, stream>>>(A, Bt, bias, Cf, Ch, Mg, Ng, Kg);
      } else {
        if (Ch) gemm_wmma<false, true,  false><<<grid, blk, 0, stream>>>(A, Bt, bias, Cf, Ch, Mg, Ng, Kg);
        else    gemm_wmma<false, false, false><<<grid, blk, 0, stream>>>(A, Bt, bias, Cf, Ch, Mg, Ng, Kg);
      }
    }
  };

  // ---- Phase 0: weights -> f16, transposed to [N,K] ----
  cvtT(fus_w1, fus_w1t, 2048, 2048);
  cvtT(fus_w2, fus_w2t, 2048, 512);
  for (int l = 0; l < L_; ++l) {
    size_t o1 = (size_t)l * H_ * H_;
    cvtT(qw + o1, qwt + o1, H_, H_);
    cvtT(kw + o1, kwt + o1, H_, H_);
    cvtT(vw + o1, vwt + o1, H_, H_);
    cvtT(rw + o1, rwt + o1, H_, H_);
    cvtT(fw + o1, fwt + o1, H_, H_);
    cvtT(ffn_w1 + (size_t)l * H_ * FF_, ffn_w1t + (size_t)l * H_ * FF_, H_, FF_);
    cvtT(ffn_w2 + (size_t)l * FF_ * H_, ffn_w2t + (size_t)l * FF_ * H_, FF_, H_);
  }

  // ---- Phase 1: rel_pos = relu(pe4 @ fus_w1 + b1) @ fus_w2 + b2  (chunked,
  //      A/B tiles staged via the Tensor Data Mover) ----
  for (int c = 0; c < R / CH; ++c) {
    int base = c * CH;
    int nel  = CH * 4 * H_;
    pe4_build<<<dim3((unsigned)(nel / 256)), dim3(256), 0, stream>>>(
        pos_s, pos_e, pe_table, pe4_h, base, nel);
    gemm(pe4_h, fus_w1t, fus_b1, nullptr, h_h, CH, 4 * H_, 4 * H_, true, true);
    gemm(h_h, fus_w2t, fus_b2, nullptr, relpos_h + (size_t)base * H_, CH, H_, 4 * H_, false, true);
  }

  // ---- x = inp ----
  (void)hipMemcpyAsync(xbuf, inp, (size_t)M * H_ * 4, hipMemcpyDeviceToDevice, stream);

  // ---- Phase 2: transformer layers ----
  for (int l = 0; l < L_; ++l) {
    size_t wo = (size_t)l * H_ * H_;
    cvt(xbuf, x_h, (size_t)M * H_);
    gemm(x_h, qwt + wo, qb + l * H_, qbuf, nullptr, M, H_, H_, false, false);
    gemm(x_h, kwt + wo, kb + l * H_, kbuf, nullptr, M, H_, H_, false, false);
    gemm(x_h, vwt + wo, vb + l * H_, vbuf, nullptr, M, H_, H_, false, false);
    gemm(relpos_h, rwt + wo, rb + l * H_, nullptr, r_h, R, H_, H_, false, true);

    scores_kernel<<<dim3((unsigned)((B_ * NH_ * S_ * S_) / 256)), dim3(256), 0, stream>>>(
        qbuf, kbuf, r_h, u + l * NH_ * D_, vp + l * NH_ * D_, sen_len, lat_len, scores);
    softmax_kernel<<<dim3((unsigned)(B_ * NH_ * S_)), dim3(192), 0, stream>>>(scores);
    attnv_kernel<<<dim3((unsigned)((M * H_) / 256)), dim3(256), 0, stream>>>(scores, vbuf, obuf);

    cvt(obuf, o_h, (size_t)M * H_);
    gemm(o_h, fwt + wo, fb + l * H_, projbuf, nullptr, M, H_, H_, false, false);
    add_ln_kernel<<<dim3((unsigned)M), dim3(256), 0, stream>>>(
        xbuf, projbuf, ln1_g + l * H_, ln1_b + l * H_, xbuf);

    cvt(xbuf, x_h, (size_t)M * H_);
    gemm(x_h, ffn_w1t + (size_t)l * H_ * FF_, ffn_b1 + l * FF_, nullptr, ff1_h, M, FF_, H_, true, false);
    gemm(ff1_h, ffn_w2t + (size_t)l * FF_ * H_, ffn_b2 + l * H_, ff2buf, nullptr, M, H_, FF_, false, false);
    float* dst = (l == L_ - 1) ? (float*)d_out : xbuf;
    add_ln_kernel<<<dim3((unsigned)M), dim3(256), 0, stream>>>(
        xbuf, ff2buf, ln2_g + l * H_, ln2_b + l * H_, dst);
  }
}
